// ToeplitzLinear_76081050681596
// MI455X (gfx1250) — matrix operand order repaired and checked by reading, not compiled, not run on hardware
//
#include <hip/hip_runtime.h>
#include <hip/hip_bf16.h>

// ---------------------------------------------------------------------------
// Toeplitz linear:  y[b,n] = sum_k x[b,k] * vals[4095 - n + k]
//   x: [8192, 4096] f32, vals: [8191] f32, y: [8192, 4096] f32
// GEMM M=8192, N=4096, K=4096 done with v_wmma_f32_16x16x32_bf16.
// B-operand (Toeplitz) is synthesized from a 16KB bf16 copy of vals in LDS.
// ---------------------------------------------------------------------------

#define BATCH_M   8192
#define FEAT_N    4096
#define FEAT_K    4096
#define NVALS     8191

#define BM 128
#define BN 128
#define BK 32
#define LDA_PAD 40          // LDS row stride for A tile in bf16 elements

typedef __attribute__((ext_vector_type(16))) __bf16 v16bf;
typedef __attribute__((ext_vector_type(8)))  float  v8f;

struct Frag32B { unsigned int u[8]; };               // 32 bytes == v16bf

static __device__ __forceinline__ unsigned short f32_to_bf16_rne(float f) {
    unsigned int u = __float_as_uint(f);
    // round-to-nearest-even into bf16
    unsigned int r = u + 0x7FFFu + ((u >> 16) & 1u);
    return (unsigned short)(r >> 16);
}

__global__ __launch_bounds__(256)
void toeplitz_wmma_bf16(const float* __restrict__ x,
                        const float* __restrict__ vals,
                        float* __restrict__ y)
{
    __shared__ unsigned short s_vals[8192];          // bf16 copy of vals (16 KB)
    __shared__ unsigned short s_a[BM * LDA_PAD];     // bf16 A tile   (10 KB)

    const int tid  = threadIdx.x;
    const int lane = tid & 31;
    const int wave = tid >> 5;
    const int wm   = wave & 3;      // wave M index: 0..3  (each covers 32 rows)
    const int wn   = wave >> 2;     // wave N index: 0..1  (each covers 64 cols)
    const int half = lane >> 4;     // lane group (0: lanes 0-15, 1: lanes 16-31)
    const int l16  = lane & 15;

    const int block_n = blockIdx.x * BN;   // fast-varying: reuse x tile in L2
    const int block_m = blockIdx.y * BM;

    // ---- stage vals -> LDS as bf16 (once per block) -------------------------
    for (int i = tid; i < NVALS; i += 256)
        s_vals[i] = f32_to_bf16_rne(vals[i]);
    if (tid == 0) s_vals[NVALS] = 0;

    // ---- accumulators: 2 (M-tiles) x 4 (N-tiles) of 16x16 f32 ---------------
    v8f acc[2][4];
    #pragma unroll
    for (int tm = 0; tm < 2; ++tm)
        #pragma unroll
        for (int tn = 0; tn < 4; ++tn)
            acc[tm][tn] = (v8f){0.f,0.f,0.f,0.f,0.f,0.f,0.f,0.f};

    // ---- K loop -------------------------------------------------------------
    for (int k0 = 0; k0 < FEAT_K; k0 += BK) {
        __syncthreads();   // protect s_a WAR (also covers s_vals on first iter)

        // stage A tile: x[block_m .. +127, k0 .. +31]  f32 -> bf16 into LDS
        // 4096 floats / 256 threads = 4 float4 per thread
        #pragma unroll
        for (int i = 0; i < 4; ++i) {
            const int idx = tid + i * 256;           // float4 index 0..1023
            const int r   = idx >> 3;                // row within tile (8 f4/row)
            const int c   = (idx & 7) * 4;           // col within tile
            const float4 v = *reinterpret_cast<const float4*>(
                &x[(size_t)(block_m + r) * FEAT_K + k0 + c]);
            const unsigned int p0 =
                ((unsigned int)f32_to_bf16_rne(v.y) << 16) | f32_to_bf16_rne(v.x);
            const unsigned int p1 =
                ((unsigned int)f32_to_bf16_rne(v.w) << 16) | f32_to_bf16_rne(v.z);
            unsigned int* dst =
                reinterpret_cast<unsigned int*>(&s_a[r * LDA_PAD + c]);
            dst[0] = p0;
            dst[1] = p1;
        }

        // prefetch next K-step's A tile into caches (global_prefetch_b8)
        if (k0 + BK < FEAT_K) {
            const int r = tid >> 1;                  // 128 rows / 2 threads each
            __builtin_prefetch(
                &x[(size_t)(block_m + r) * FEAT_K + (k0 + BK) + (tid & 1) * 16],
                0, 1);
        }

        __syncthreads();

        // ---- build A fragments (16-bit A 16x32 layout, ISA 7.12.2) ----------
        // VGPR v (uint pair): k = (v<4 ? 2v : 16 + 2(v-4)) + 8*half, row = l16
        Frag32B afrag[2];
        #pragma unroll
        for (int tm = 0; tm < 2; ++tm) {
            const int m    = wm * 32 + tm * 16 + l16;
            const int base = m * LDA_PAD;
            #pragma unroll
            for (int v = 0; v < 8; ++v) {
                const int k = ((v < 4) ? (2 * v) : (16 + 2 * (v - 4))) + 8 * half;
                afrag[tm].u[v] = *reinterpret_cast<const unsigned int*>(
                    &s_a[base + k]);                 // 4B aligned (base, k even)
            }
        }

        // ---- synthesize B fragments from vals (Toeplitz) --------------------
        // B 32x16 (KxN) 16-bit layout: lane col n = l16;
        // VGPR v: k = (v<4 ? 2v : 16 + 2(v-4)) + 8*half (2 K per VGPR).
        Frag32B bfrag[4];
        #pragma unroll
        for (int tn = 0; tn < 4; ++tn) {
            const int n     = block_n + wn * 64 + tn * 16 + l16;
            const int vbase = (FEAT_N - 1) - n + k0 + half * 8;  // in [0, 8190]
            #pragma unroll
            for (int v = 0; v < 8; ++v) {
                const int koff = (v < 4) ? (2 * v) : (16 + 2 * (v - 4));
                const unsigned int lo = s_vals[vbase + koff];
                const unsigned int hi = s_vals[vbase + koff + 1];
                bfrag[tn].u[v] = lo | (hi << 16);
            }
        }

        // ---- 8 WMMAs per wave per K-step ------------------------------------
        #pragma unroll
        for (int tm = 0; tm < 2; ++tm) {
            const v16bf a = __builtin_bit_cast(v16bf, afrag[tm]);
            #pragma unroll
            for (int tn = 0; tn < 4; ++tn) {
                const v16bf b = __builtin_bit_cast(v16bf, bfrag[tn]);
                acc[tm][tn] = __builtin_amdgcn_wmma_f32_16x16x32_bf16(
                    /*neg_a=*/false, a, /*neg_b=*/false, b,
                    /*c_mod=*/(short)0, acc[tm][tn],
                    /*reuse_a=*/false, /*reuse_b=*/false);
            }
        }
    }

    // ---- writeback: C/D 16x16 f32 layout (m = vgpr + 8*half, n = l16) -------
    #pragma unroll
    for (int tm = 0; tm < 2; ++tm) {
        const int row0 = block_m + wm * 32 + tm * 16 + 8 * half;
        #pragma unroll
        for (int tn = 0; tn < 4; ++tn) {
            const int col = block_n + wn * 64 + tn * 16 + l16;
            #pragma unroll
            for (int v = 0; v < 8; ++v) {
                y[(size_t)(row0 + v) * FEAT_N + col] = acc[tm][tn][v];
            }
        }
    }
}

extern "C" void kernel_launch(void* const* d_in, const int* in_sizes, int n_in,
                              void* d_out, int out_size, void* d_ws, size_t ws_size,
                              hipStream_t stream) {
    (void)in_sizes; (void)n_in; (void)out_size; (void)d_ws; (void)ws_size;
    const float* x    = (const float*)d_in[0];   // [8192, 4096] f32
    const float* vals = (const float*)d_in[1];   // [8191] f32
    float*       y    = (float*)d_out;           // [8192, 4096] f32

    dim3 grid(FEAT_N / BN, BATCH_M / BM);        // (32, 64)
    dim3 block(256);                             // 8 wave32s
    toeplitz_wmma_bf16<<<grid, block, 0, stream>>>(x, vals, y);
}
